// HelixAttention_53876069761175
// MI455X (gfx1250) — matrix operand order repaired and checked by reading, not compiled, not run on hardware
//
#include <hip/hip_runtime.h>
#include <hip/hip_bf16.h>

typedef __attribute__((ext_vector_type(16))) _Float16 v16h;
typedef __attribute__((ext_vector_type(8)))  _Float16 v8h;
typedef __attribute__((ext_vector_type(8)))  float    v8f;
typedef __attribute__((ext_vector_type(4)))  int      v4i;

#define BQ  4
#define SQ  2048
#define DQ  2048
#define HQ  16
#define DHQ 128
#define MQ  (BQ*SQ)

static __device__ __forceinline__ v16h join8(v8h lo, v8h hi) {
  v16h r;
#pragma unroll
  for (int i = 0; i < 8; ++i) { r[i] = lo[i]; r[8 + i] = hi[i]; }
  return r;
}

static __device__ __forceinline__ v8f wmma16(v16h a, v16h b, v8f c) {
  // D = A(16x32 f16) * B(32x16 f16) + C(16x16 f32)
  return __builtin_amdgcn_wmma_f32_16x16x32_f16(false, a, false, b, (short)0, c,
                                                false, false);
}

// ---- async global -> LDS (16B per lane), ASYNCcnt tracked -----------------
typedef __attribute__((address_space(1))) v4i gv4i;
typedef __attribute__((address_space(3))) v4i lv4i;

static __device__ __forceinline__ void async_ld128(const void* g, void* l) {
#if __has_builtin(__builtin_amdgcn_global_load_async_to_lds_b128)
  __builtin_amdgcn_global_load_async_to_lds_b128((gv4i*)g, (lv4i*)l, 0, 0);
#else
  unsigned loff =
      (unsigned)(unsigned long long)(__attribute__((address_space(3))) char*)l;
  unsigned long long ga = (unsigned long long)g;
  asm volatile("global_load_async_to_lds_b128 %0, %1, off"
               :: "v"(loff), "v"(ga) : "memory");
#endif
}

static __device__ __forceinline__ void wait_async0() {
#if __has_builtin(__builtin_amdgcn_s_wait_asynccnt)
  __builtin_amdgcn_s_wait_asynccnt(0);
#else
  asm volatile("s_wait_asynccnt 0" ::: "memory");
#endif
}

// ---------------------------------------------------------------------------
// f32 -> f16 cast (x and the four weight matrices)
// ---------------------------------------------------------------------------
__global__ void cast_f32_f16(const float* __restrict__ src,
                             _Float16* __restrict__ dst, size_t n) {
  size_t i = (size_t)blockIdx.x * blockDim.x + threadIdx.x;
  size_t stride = (size_t)gridDim.x * blockDim.x;
  for (; i < n; i += stride) dst[i] = (_Float16)src[i];
}

// ---------------------------------------------------------------------------
// C[m,n] = sum_k X[m,k] * W[n,k] + bias[n], then per-channel gate:
//   mode 0: *cos(rot[n])  (Q)   mode 1: *sin(rot[n])  (K)
//   mode 2: none -> f16   (V)   mode 3: none -> f32   (final out proj)
// Block tile 128x128, 8 waves, each wave 32x64 (2x4 WMMA tiles), BK=32.
// Double-buffered LDS fed by global_load_async_to_lds_b128.
// ---------------------------------------------------------------------------
#define PBM 128
#define PBN 128
#define PBK 32
#define PLDA (PBK + 8)

__global__ __launch_bounds__(256) void proj_gemm(
    const _Float16* __restrict__ X,   // [MQ, DQ] row-major
    const _Float16* __restrict__ W,   // [DQ, DQ] row-major ([n][k])
    const float* __restrict__ bias,   // [DQ]
    const float* __restrict__ rot,    // [DQ] helix_rotation (modes 0/1)
    _Float16* __restrict__ outH,      // f16 out (modes 0..2)
    float* __restrict__ outF,         // f32 out (mode 3)
    int mode) {
  __shared__ _Float16 As[2][PBM][PLDA];  // 2 x 10 KB
  __shared__ _Float16 Bs[2][PBN][PLDA];  // 2 x 10 KB

  const int tid  = threadIdx.x;
  const int wv   = tid >> 5;
  const int lane = tid & 31;
  const int hl   = lane >> 4;
  const int ln   = lane & 15;
  const int m0   = blockIdx.x * PBM;
  const int n0   = blockIdx.y * PBN;
  const int wm   = (wv & 3) * 32;   // 4 waves over M
  const int wn   = (wv >> 2) * 64;  // 2 waves over N

  // issue one 128x32 A tile + 128x32 B tile as async 16B chunks (4/thread)
  auto issue = [&](int k0, int buf) {
#pragma unroll
    for (int u = 0; u < 2; ++u) {
      const int q = tid + u * 256;            // 512 chunks of 16B
      const int r = q >> 2, c = (q & 3) * 8;
      async_ld128(X + (size_t)(m0 + r) * DQ + k0 + c, &As[buf][r][c]);
    }
#pragma unroll
    for (int u = 0; u < 2; ++u) {
      const int q = tid + u * 256;
      const int r = q >> 2, c = (q & 3) * 8;
      async_ld128(W + (size_t)(n0 + r) * DQ + k0 + c, &Bs[buf][r][c]);
    }
  };

  v8f acc[2][4] = {};
  int cur = 0;
  issue(0, 0);

  for (int k0 = 0; k0 < DQ; k0 += PBK) {
    wait_async0();
    __syncthreads();
    if (k0 + PBK < DQ) issue(k0 + PBK, cur ^ 1);

    v16h af[2], bf[4];
#pragma unroll
    for (int i = 0; i < 2; ++i) {  // A frag: lane=row, K split by half
      const _Float16* p = &As[cur][wm + i * 16 + ln][0];
      af[i] = join8(*(const v8h*)(p + 16 * hl), *(const v8h*)(p + 16 * hl + 8));
    }
#pragma unroll
    for (int j = 0; j < 4; ++j) {  // B frag: lane=col, K = 16*hl..+15
      const _Float16* p = &Bs[cur][wn + j * 16 + ln][16 * hl];
      bf[j] = join8(*(const v8h*)p, *(const v8h*)(p + 8));
    }
#pragma unroll
    for (int i = 0; i < 2; ++i)
#pragma unroll
      for (int j = 0; j < 4; ++j) acc[i][j] = wmma16(af[i], bf[j], acc[i][j]);
    cur ^= 1;
  }

  // epilogue: C layout -> VGPR r holds row (r + 8*hl), lane ln = column
#pragma unroll
  for (int j = 0; j < 4; ++j) {
    const int n = n0 + wn + j * 16 + ln;
    float g = 1.0f;
    if (mode == 0) g = __cosf(rot[n]);
    else if (mode == 1) g = __sinf(rot[n]);
    const float bb = bias[n];
#pragma unroll
    for (int i = 0; i < 2; ++i) {
#pragma unroll
      for (int r = 0; r < 8; ++r) {
        const int m = m0 + wm + i * 16 + r + 8 * hl;
        const float val = (acc[i][j][r] + bb) * g;
        if (mode == 3) outF[(size_t)m * DQ + n] = val;
        else           outH[(size_t)m * DQ + n] = (_Float16)val;
      }
    }
  }
}

// ---------------------------------------------------------------------------
// Flash attention per (b, h, 64-query block). 4 waves x 16 queries.
// Key blocks of 32, double-buffered: K async to LDS, V via regs (transposed).
// ---------------------------------------------------------------------------
#define TK 32
#define KLD (DHQ + 8)
#define VLD (TK + 8)

__global__ __launch_bounds__(128) void helix_attn(
    const _Float16* __restrict__ Q, const _Float16* __restrict__ K,
    const _Float16* __restrict__ V, const float* __restrict__ spiral,
    _Float16* __restrict__ O) {
  __shared__ _Float16 Ks[2][TK][KLD];     // 2 x 8.7 KB
  __shared__ _Float16 Vt[2][DHQ][VLD];    // 2 x 10.2 KB (transposed V)
  __shared__ _Float16 Ps[4][16][VLD];     // per-wave P transpose, 5.1 KB

  const int tid  = threadIdx.x;
  const int wv   = tid >> 5;
  const int lane = tid & 31;
  const int hl   = lane >> 4;
  const int ln   = lane & 15;

  const int nsb  = SQ / 64;
  const int sblk = blockIdx.x % nsb;
  const int bh   = blockIdx.x / nsb;
  const int h    = bh % HQ;
  const int b    = bh / HQ;
  const int s0   = sblk * 64 + wv * 16;           // this wave's 16 queries
  const size_t rowbase = (size_t)b * SQ * DQ + (size_t)h * DHQ;

  auto issueK = [&](int t0, int buf) {            // 512 chunks, 4/thread
#pragma unroll
    for (int u = 0; u < 4; ++u) {
      const int q = tid * 4 + u;
      const int r = q >> 4, c = (q & 15) * 8;
      async_ld128(K + rowbase + (size_t)(t0 + r) * DQ + c, &Ks[buf][r][c]);
    }
  };
  const int vr_r = tid >> 2, vr_c = (tid & 3) * 32;
  auto loadV = [&](int t0, v8h* vr) {
    const _Float16* vp = V + rowbase + (size_t)(t0 + vr_r) * DQ + vr_c;
    vr[0] = *(const v8h*)vp;        vr[1] = *(const v8h*)(vp + 8);
    vr[2] = *(const v8h*)(vp + 16); vr[3] = *(const v8h*)(vp + 24);
  };
  auto storeVt = [&](const v8h* vr, int buf) {    // transposed: Vt[dh][t]
#pragma unroll
    for (int e = 0; e < 8; ++e) {
      Vt[buf][vr_c + e][vr_r]      = vr[0][e];
      Vt[buf][vr_c + 8 + e][vr_r]  = vr[1][e];
      Vt[buf][vr_c + 16 + e][vr_r] = vr[2][e];
      Vt[buf][vr_c + 24 + e][vr_r] = vr[3][e];
    }
  };

  // Q A-fragments for the wave's 16 rows, full DH=128 (4 chunks of K=32)
  v16h qf[4];
  {
    const _Float16* qp = Q + rowbase + (size_t)(s0 + ln) * DQ;
#pragma unroll
    for (int kk = 0; kk < 4; ++kk) {
      v8h lo = *(const v8h*)(qp + kk * 32 + 16 * hl);
      v8h hi = *(const v8h*)(qp + kk * 32 + 16 * hl + 8);
      qf[kk] = join8(lo, hi);
    }
  }
  const float scale = spiral[h] * 0.0883883476483184f;  // 1/sqrt(128)

  v8f o[8] = {};
  float mrow[8], lrow[8];
#pragma unroll
  for (int r = 0; r < 8; ++r) { mrow[r] = -3.0e38f; lrow[r] = 0.0f; }

  int cur = 0;
  issueK(0, 0);
  {
    v8h vr0[4];
    loadV(0, vr0);
    storeVt(vr0, 0);
  }

  for (int t0 = 0; t0 < SQ; t0 += TK) {
    wait_async0();          // this wave's K-tile chunks landed in LDS
    __syncthreads();        // everyone's K/V tile visible; prev compute done

    const bool more = (t0 + TK) < SQ;
    v8h vr[4];
    if (more) {
      issueK(t0 + TK, cur ^ 1);   // async, overlaps with score WMMAs
      loadV(t0 + TK, vr);         // latency hides under scores/softmax
      if (t0 + 2 * TK < SQ)       // -> global_prefetch_b8
        __builtin_prefetch(V + rowbase + (size_t)(t0 + 2 * TK + vr_r) * DQ + vr_c,
                           0, 1);
    }

    // scores: two 16x16 tiles covering keys [t0, t0+32)
    v8f sc0 = {}, sc1 = {};
#pragma unroll
    for (int kk = 0; kk < 4; ++kk) {
      const _Float16* p0 = &Ks[cur][ln][kk * 32 + 16 * hl];
      const _Float16* p1 = &Ks[cur][16 + ln][kk * 32 + 16 * hl];
      v16h b0 = join8(*(const v8h*)p0, *(const v8h*)(p0 + 8));
      v16h b1 = join8(*(const v8h*)p1, *(const v8h*)(p1 + 8));
      sc0 = wmma16(qf[kk], b0, sc0);
      sc1 = wmma16(qf[kk], b1, sc1);
    }

    // online softmax (rows live across the 16-lane half, one row per VGPR)
    float corr[8];
#pragma unroll
    for (int r = 0; r < 8; ++r) {
      float v0 = sc0[r] * scale;
      float v1 = sc1[r] * scale;
      float mx = fmaxf(v0, v1);
      mx = fmaxf(mx, __shfl_xor(mx, 1, 32));
      mx = fmaxf(mx, __shfl_xor(mx, 2, 32));
      mx = fmaxf(mx, __shfl_xor(mx, 4, 32));
      mx = fmaxf(mx, __shfl_xor(mx, 8, 32));
      float mn = fmaxf(mrow[r], mx);
      float p0 = __expf(v0 - mn);
      float p1 = __expf(v1 - mn);
      float cr = __expf(mrow[r] - mn);
      float rs = p0 + p1;
      rs += __shfl_xor(rs, 1, 32);
      rs += __shfl_xor(rs, 2, 32);
      rs += __shfl_xor(rs, 4, 32);
      rs += __shfl_xor(rs, 8, 32);
      lrow[r] = lrow[r] * cr + rs;
      mrow[r] = mn;
      corr[r] = cr;
      Ps[wv][r + 8 * hl][ln]      = (_Float16)p0;   // C layout -> LDS
      Ps[wv][r + 8 * hl][16 + ln] = (_Float16)p1;
    }
#pragma unroll
    for (int n = 0; n < 8; ++n)
#pragma unroll
      for (int r = 0; r < 8; ++r) o[n][r] *= corr[r];

    if (more) storeVt(vr, cur ^ 1);  // other buffer; safe vs everyone's reads

    asm volatile("s_wait_dscnt 0" ::: "memory");  // Ps visible to own wave

    // P A-fragment (16x32), then PV over the 8 dh tiles
    v16h pf;
    {
      const _Float16* p = &Ps[wv][ln][0];
      pf = join8(*(const v8h*)(p + 16 * hl), *(const v8h*)(p + 16 * hl + 8));
    }
#pragma unroll
    for (int n = 0; n < 8; ++n) {
      const _Float16* p = &Vt[cur][n * 16 + ln][16 * hl];
      v16h bv = join8(*(const v8h*)p, *(const v8h*)(p + 8));
      o[n] = wmma16(pf, bv, o[n]);
    }
    cur ^= 1;
  }

  // normalize and write f16 attention output [B,S,D]
#pragma unroll
  for (int n = 0; n < 8; ++n) {
#pragma unroll
    for (int r = 0; r < 8; ++r) {
      const int m = r + 8 * hl;
      const float val = o[n][r] / lrow[r];
      O[rowbase + (size_t)(s0 + m) * DQ + n * 16 + ln] = (_Float16)val;
    }
  }
}

// ---------------------------------------------------------------------------
extern "C" void kernel_launch(void* const* d_in, const int* in_sizes, int n_in,
                              void* d_out, int out_size, void* d_ws,
                              size_t ws_size, hipStream_t stream) {
  const float* x      = (const float*)d_in[0];
  const float* Wq     = (const float*)d_in[1];
  const float* bq     = (const float*)d_in[2];
  const float* Wk     = (const float*)d_in[3];
  const float* bk     = (const float*)d_in[4];
  const float* Wv     = (const float*)d_in[5];
  const float* bv     = (const float*)d_in[6];
  const float* Wo     = (const float*)d_in[7];
  const float* bo     = (const float*)d_in[8];
  const float* spiral = (const float*)d_in[9];
  const float* helix  = (const float*)d_in[10];

  char* ws = (char*)d_ws;
  const size_t MD = (size_t)MQ * DQ;   // 16.78M elems
  const size_t DD = (size_t)DQ * DQ;   //  4.19M elems
  _Float16* xh  = (_Float16*)(ws);                       // 32 MB
  _Float16* Qh  = (_Float16*)(ws + MD * 2);              // 32 MB
  _Float16* Kh  = (_Float16*)(ws + MD * 4);              // 32 MB
  _Float16* Vh  = (_Float16*)(ws + MD * 6);              // 32 MB
  _Float16* Wqh = (_Float16*)(ws + MD * 8);              //  8 MB
  _Float16* Wkh = (_Float16*)(ws + MD * 8 + DD * 2);
  _Float16* Wvh = (_Float16*)(ws + MD * 8 + DD * 4);
  _Float16* Woh = (_Float16*)(ws + MD * 8 + DD * 6);
  _Float16* Oh  = xh;  // x no longer needed once Q/K/V exist -> reuse

  cast_f32_f16<<<2048, 256, 0, stream>>>(x, xh, MD);
  cast_f32_f16<<<512, 256, 0, stream>>>(Wq, Wqh, DD);
  cast_f32_f16<<<512, 256, 0, stream>>>(Wk, Wkh, DD);
  cast_f32_f16<<<512, 256, 0, stream>>>(Wv, Wvh, DD);
  cast_f32_f16<<<512, 256, 0, stream>>>(Wo, Woh, DD);

  dim3 pg(MQ / PBM, DQ / PBN);  // (64, 16)
  proj_gemm<<<pg, 256, 0, stream>>>(xh, Wqh, bq, helix, Qh, nullptr, 0);
  proj_gemm<<<pg, 256, 0, stream>>>(xh, Wkh, bk, helix, Kh, nullptr, 1);
  proj_gemm<<<pg, 256, 0, stream>>>(xh, Wvh, bv, nullptr, Vh, nullptr, 2);

  helix_attn<<<BQ * HQ * (SQ / 64), 128, 0, stream>>>(Qh, Kh, Vh, spiral, Oh);

  proj_gemm<<<pg, 256, 0, stream>>>(Oh, Woh, bo, nullptr, nullptr,
                                    (float*)d_out, 3);
}
